// ExpKernelAttention_8589934592226
// MI455X (gfx1250) — matrix operand order repaired and checked by reading, not compiled, not run on hardware
//
#include <hip/hip_runtime.h>
#include <math.h>

// Problem constants (from reference): B=4, H=16, S=4096, D=64
#define Bc 4
#define Hc 16
#define Sc 4096
#define Dc 64
#define BH (Bc*Hc)          // 64 heads
#define NCH1 8              // S-chunks for stats kernel
#define NCH2 4              // S-chunks for KV-GEMM kernel

// Workspace layout (in floats)
#define WS_PMAX  0                          // BH*NCH1*64 = 32768
#define WS_PSUM  32768                      // 32768
#define WS_CMAX  65536                      // BH*64 = 4096
#define WS_CINV  69632                      // 4096
#define WS_DOTP  73728                      // BH*NCH2*64*64 = 1048576
#define WS_DOTF  1122304                    // BH*64*64 = 262144  (total ~5.5 MB)

typedef __attribute__((ext_vector_type(16))) _Float16 v16h;
typedef __attribute__((ext_vector_type(8)))  float    v8f;

// ---------------- Kernel 1: per-column (axis=S) online max/sumexp of K*mask ----
__global__ void __launch_bounds__(256)
k_colstats(const float* __restrict__ Kg, const float* __restrict__ mask,
           float* __restrict__ ws) {
  int bh = blockIdx.x >> 3;
  int ch = blockIdx.x & 7;
  int t  = threadIdx.x;
  int d  = t & 63;
  int sub = t >> 6;                       // 0..3
  const int rows = Sc / NCH1;             // 512
  const float* Kp = Kg + (size_t)bh * Sc * Dc;
  const float* mp = mask + (size_t)(bh >> 4) * Sc + ch * rows;

  float m = -INFINITY, sum = 0.f;
  for (int i = 0; i < rows / 4; ++i) {
    int s = ch * rows + sub + 4 * i;
    float x = Kp[(size_t)s * Dc + d] * mp[sub + 4 * i];
    float mn = fmaxf(m, x);
    sum = sum * __expf(m - mn) + __expf(x - mn);
    m = mn;
  }
  __shared__ float smax[256], ssum[256];
  smax[t] = m; ssum[t] = sum;
  __syncthreads();
  if (t < 64) {
    float M = smax[t];
    for (int j = 1; j < 4; ++j) M = fmaxf(M, smax[t + 64 * j]);
    float SS = 0.f;
    for (int j = 0; j < 4; ++j) SS += ssum[t + 64 * j] * __expf(smax[t + 64 * j] - M);
    ws[WS_PMAX + ((size_t)bh * NCH1 + ch) * 64 + t] = M;
    ws[WS_PSUM + ((size_t)bh * NCH1 + ch) * 64 + t] = SS;
  }
}

// ---------------- Kernel 2: merge chunk stats -> colMax, colInv ---------------
__global__ void __launch_bounds__(64)
k_merge(float* __restrict__ ws) {
  int bh = blockIdx.x;
  int t  = threadIdx.x;   // = d
  float M = -INFINITY;
  for (int c = 0; c < NCH1; ++c)
    M = fmaxf(M, ws[WS_PMAX + ((size_t)bh * NCH1 + c) * 64 + t]);
  float SS = 0.f;
  for (int c = 0; c < NCH1; ++c)
    SS += ws[WS_PSUM + ((size_t)bh * NCH1 + c) * 64 + t] *
          __expf(ws[WS_PMAX + ((size_t)bh * NCH1 + c) * 64 + t] - M);
  ws[WS_CMAX + (size_t)bh * 64 + t] = M;
  ws[WS_CINV + (size_t)bh * 64 + t] = 1.0f / SS;
}

// ---------------- Kernel 3: dot_partial = Ks^T @ Vm via WMMA f16 --------------
// 128 threads = 4 waves; wave w owns d-rows [16w,16w+16), all 64 e-columns.
__global__ void __launch_bounds__(128)
k_kv(const float* __restrict__ Kg, const float* __restrict__ Vg,
     const float* __restrict__ mask, float* __restrict__ ws) {
  int bh = blockIdx.x >> 2;
  int ch = blockIdx.x & 3;
  int t  = threadIdx.x;
  int w = t >> 5, l = t & 31, m16 = l & 15, half = l >> 4;
  const float* Kp = Kg + (size_t)bh * Sc * Dc;
  const float* Vp = Vg + (size_t)bh * Sc * Dc;
  const float* mp = mask + (size_t)(bh >> 4) * Sc;

  int dlane = 16 * w + m16;
  float cmax = ws[WS_CMAX + (size_t)bh * 64 + dlane];
  float cinv = ws[WS_CINV + (size_t)bh * 64 + dlane];

  __shared__ float    Kt[64 * 33];   // [d][s], padded stride 33
  __shared__ _Float16 Vt[64 * 34];   // [e][s], padded stride 34 (pairs stay adjacent)

  v8f acc[4];
  #pragma unroll
  for (int et = 0; et < 4; ++et)
    #pragma unroll
    for (int i = 0; i < 8; ++i) acc[et][i] = 0.f;

  int sBase = ch * (Sc / NCH2);      // 1024 rows per chunk
  for (int it = 0; it < (Sc / NCH2) / 32; ++it) {   // 32 iterations
    int s0 = sBase + 32 * it;
    __syncthreads();
    for (int j = 0; j < 16; ++j) {                  // 32x64 tile, transposed into LDS
      int idx = t + 128 * j;
      int r = idx >> 6, c = idx & 63;
      float mv = mp[s0 + r];
      Kt[c * 33 + r] = Kp[(size_t)(s0 + r) * Dc + c] * mv;
      Vt[c * 34 + r] = (_Float16)(Vp[(size_t)(s0 + r) * Dc + c] * mv);
    }
    __syncthreads();

    // A fragment: 16x32, rows = d, K = s  (16-bit A layout, ISA 7.12.2)
    v16h a;
    #pragma unroll
    for (int v = 0; v < 8; ++v) {
      int kk = (v & 3) * 2 + ((v >> 2) << 4) + (half << 3);
      a[2 * v]     = (_Float16)(__expf(Kt[dlane * 33 + kk]     - cmax) * cinv);
      a[2 * v + 1] = (_Float16)(__expf(Kt[dlane * 33 + kk + 1] - cmax) * cinv);
    }
    #pragma unroll
    for (int et = 0; et < 4; ++et) {
      int e = 16 * et + m16;
      v16h bf;                                       // B fragment: 32x16, rows = s
      #pragma unroll
      for (int v = 0; v < 8; ++v) {
        int kk = (v & 3) * 2 + ((v >> 2) << 4) + (half << 3);
        bf[2 * v]     = Vt[e * 34 + kk];
        bf[2 * v + 1] = Vt[e * 34 + kk + 1];
      }
      acc[et] = __builtin_amdgcn_wmma_f32_16x16x32_f16(
          false, a, false, bf, (short)0, acc[et], false, false);
    }
  }

  float* dp = ws + WS_DOTP + ((size_t)bh * NCH2 + ch) * 4096;
  #pragma unroll
  for (int et = 0; et < 4; ++et)
    #pragma unroll
    for (int r = 0; r < 8; ++r) {
      int dd = 16 * w + r + 8 * half;                // C/D layout: M = vgpr + 8*half
      int e  = 16 * et + m16;
      dp[dd * 64 + e] = acc[et][r];
    }
}

// ---------------- Kernel 4: deterministic reduction of dot partials -----------
__global__ void __launch_bounds__(256)
k_dotmerge(float* __restrict__ ws) {
  int bh = blockIdx.x;
  int t  = threadIdx.x;
  const float* dp = ws + WS_DOTP + (size_t)bh * NCH2 * 4096;
  float* df = ws + WS_DOTF + (size_t)bh * 4096;
  for (int j = 0; j < 16; ++j) {
    int idx = t + 256 * j;
    df[idx] = dp[idx] + dp[4096 + idx] + dp[8192 + idx] + dp[12288 + idx];
  }
}

// ---------------- Kernel 5: attn = rowsoftmax(Q*mask) @ dot via WMMA ----------
// 256 threads = 8 waves; each wave owns a 16-row s-tile of a 128-row block.
__global__ void __launch_bounds__(256)
k_attn(const float* __restrict__ Qg, const float* __restrict__ mask,
       const float* __restrict__ ws, float* __restrict__ out) {
  int bh   = blockIdx.x >> 5;
  int sblk = blockIdx.x & 31;
  int sBase = sblk * 128;
  int t = threadIdx.x;
  int w = t >> 5, l = t & 31, m16 = l & 15, half = l >> 4;

  __shared__ float    Qraw[128 * 65];   // padded f32 staging (reused for output)
  __shared__ _Float16 Qs[128 * 66];     // softmaxed Q, f16, padded
  __shared__ _Float16 dotT[64 * 66];    // dot transposed [e][d], f16, padded
  __shared__ float    rinv[128];

  const float* Qp  = Qg + (size_t)bh * Sc * Dc;
  const float* mp  = mask + (size_t)(bh >> 4) * Sc + sBase;
  const float* dot = ws + WS_DOTF + (size_t)bh * 4096;

  for (int j = 0; j < 16; ++j) {        // 64x64 dot -> transposed f16
    int idx = t + 256 * j;
    int dd = idx >> 6, e = idx & 63;
    dotT[e * 66 + dd] = (_Float16)dot[idx];
  }
  for (int j = 0; j < 32; ++j) {        // 128x64 Q tile, masked
    int idx = t + 256 * j;
    int r = idx >> 6, c = idx & 63;
    Qraw[r * 65 + c] = Qp[(size_t)(sBase + r) * Dc + c] * mp[r];
  }
  __syncthreads();

  if (t < 128) {                        // row softmax over D=64
    float M = -INFINITY;
    for (int c = 0; c < 64; ++c) M = fmaxf(M, Qraw[t * 65 + c]);
    float SS = 0.f;
    for (int c = 0; c < 64; ++c) {
      float e = __expf(Qraw[t * 65 + c] - M);
      SS += e;
      Qs[t * 66 + c] = (_Float16)e;
    }
    rinv[t] = 1.0f / SS;
  }
  __syncthreads();
  for (int j = 0; j < 32; ++j) {        // normalize
    int idx = t + 256 * j;
    int r = idx >> 6, c = idx & 63;
    Qs[r * 66 + c] = (_Float16)((float)Qs[r * 66 + c] * rinv[r]);
  }
  __syncthreads();

  v8f acc[4];
  #pragma unroll
  for (int et = 0; et < 4; ++et)
    #pragma unroll
    for (int i = 0; i < 8; ++i) acc[et][i] = 0.f;

  int rowBase = 16 * w;
  #pragma unroll
  for (int ks = 0; ks < 2; ++ks) {      // contract over D=64 in two K=32 steps
    v16h a;
    #pragma unroll
    for (int v = 0; v < 8; ++v) {
      int kk = (v & 3) * 2 + ((v >> 2) << 4) + (half << 3) + 32 * ks;
      a[2 * v]     = Qs[(rowBase + m16) * 66 + kk];
      a[2 * v + 1] = Qs[(rowBase + m16) * 66 + kk + 1];
    }
    #pragma unroll
    for (int et = 0; et < 4; ++et) {
      v16h bf;
      #pragma unroll
      for (int v = 0; v < 8; ++v) {
        int kk = (v & 3) * 2 + ((v >> 2) << 4) + (half << 3) + 32 * ks;
        bf[2 * v]     = dotT[(16 * et + m16) * 66 + kk];
        bf[2 * v + 1] = dotT[(16 * et + m16) * 66 + kk + 1];
      }
      acc[et] = __builtin_amdgcn_wmma_f32_16x16x32_f16(
          false, a, false, bf, (short)0, acc[et], false, false);
    }
  }
  __syncthreads();

  #pragma unroll
  for (int et = 0; et < 4; ++et)        // stage C tiles for coalesced store
    #pragma unroll
    for (int r = 0; r < 8; ++r) {
      int row = rowBase + r + 8 * half;
      Qraw[row * 65 + 16 * et + m16] = acc[et][r];
    }
  __syncthreads();

  float* op = out + (size_t)bh * Sc * Dc + (size_t)sBase * Dc;
  for (int j = 0; j < 32; ++j) {
    int idx = t + 256 * j;
    int r = idx >> 6, c = idx & 63;
    op[idx] = Qraw[r * 65 + c];
  }
}

extern "C" void kernel_launch(void* const* d_in, const int* in_sizes, int n_in,
                              void* d_out, int out_size, void* d_ws, size_t ws_size,
                              hipStream_t stream) {
  const float* Q    = (const float*)d_in[0];
  const float* K    = (const float*)d_in[1];
  const float* V    = (const float*)d_in[2];
  const float* mask = (const float*)d_in[3];
  float* ws  = (float*)d_ws;
  float* out = (float*)d_out;

  k_colstats<<<BH * NCH1, 256, 0, stream>>>(K, mask, ws);
  k_merge   <<<BH,        64,  0, stream>>>(ws);
  k_kv      <<<BH * NCH2, 128, 0, stream>>>(K, V, mask, ws);
  k_dotmerge<<<BH,        256, 0, stream>>>(ws);
  k_attn    <<<BH * 32,   256, 0, stream>>>(Q, mask, ws, out);
}
